// ThickCubeSimulator_20263655702557
// MI455X (gfx1250) — compile-verified, matrix operand order checked by reading, not run on hardware
//
#include <hip/hip_runtime.h>
#include <hip/hip_bf16.h>

typedef __attribute__((ext_vector_type(16))) _Float16 v16h;
typedef __attribute__((ext_vector_type(8)))  float    v8f;

#define C_KMS    299792.458f
#define F0_GHZ   230.538f
#define FOV_HALF 1000.0f
#define RES_OUT  80
// fine grid N = 160, fine freq F = 160, z-chunks of 32, f-tiles of 16

__global__ __launch_bounds__(320)
void ThickCube_wmma_kernel(const float* __restrict__ inc_p,
                           const float* __restrict__ rot_p,
                           const float* __restrict__ lb_p,
                           const float* __restrict__ vs_p,
                           const float* __restrict__ vmax_p,
                           const float* __restrict__ rturn_p,
                           const float* __restrict__ i0_p,
                           const float* __restrict__ rd_p,
                           const float* __restrict__ fc_p,
                           float* __restrict__ out)
{
    // Per (column, z): linearized Gaussian exponent coefficients (m, p) with
    //   q(f,z) = p + vlab_f * m - K_f,   K_f = h2*vlab_f^2 (per-lane constant)
    __shared__ __align__(16) float2 s_col[4 * 160];

    const int blk = blockIdx.x;          // 0..6399 : one pooled output pixel
    const int ic  = blk / 80;            // coarse i (first spatial axis)
    const int jc  = blk % 80;            // coarse j (second spatial axis)
    const int tid = threadIdx.x;

    // ---- uniform parameters (scalar loads) ----
    const float inc    = inc_p[0];
    const float rot    = rot_p[0];
    const float lb     = lb_p[0];
    const float vshift = vs_p[0];
    const float vmax   = vmax_p[0];
    const float rturn  = rturn_p[0];
    const float i0     = i0_p[0];
    const float rd     = rd_p[0];
    const float fc0    = fc_p[0];
    const float fc1    = fc_p[1];

    const float ci = cosf(inc), si = sinf(inc);
    const float cr = cosf(rot), sr = sinf(rot);

    const float dxh = (2.0f * FOV_HALF / RES_OUT) * 0.5f;   // 12.5 pc fine pixel
    const float x0  = -FOV_HALF + 0.5f * dxh;               // -993.75

    const float log2e   = 1.44269504088896340736f;
    const float s2      = lb * lb;
    const float h2      = (0.5f / s2) * log2e;              // Gaussian coeff, log2 domain
    const float invrdl2 = log2e / rd;                       // disk coeff, log2 domain

    // fine frequency axis
    const float df  = (fc1 - fc0) * 0.25f;
    const float ff0 = fc0 - 1.5f * df;

    // ---- Phase 1: per-(column,z) geometry into LDS ----
    for (int e = tid; e < 4 * 160; e += 320) {
        const int c  = e / 160;                 // column 0..3 of 2x2 block
        const int z  = e - c * 160;
        const int cx = c >> 1, cy = c & 1;

        const float gx = x0 + dxh * (float)(ic * 2 + cx);
        const float gy = x0 + dxh * (float)(jc * 2 + cy);
        const float gz = x0 + dxh * (float)z;

        const float rx = gx * cr - gy * sr;                 // x1
        const float y1 = gx * sr + gy * cr;
        const float ry = y1 * ci - gz * si;
        const float rz = y1 * si + gz * ci;

        const float rc2   = rx * rx + ry * ry;
        const float rsph  = sqrtf(rc2 + rz * rz);
        const float invrc = rc2 > 0.0f ? rsqrtf(rc2) : 0.0f;
        const float rcyl  = rc2 * invrc;

        const float vabs = vmax * 0.636619772367581343f * atanf(rcyl / rturn);
        const float vlos = -(vabs * rx * invrc) * si;       // -v_y * sin(inc)
        const float cl2  = -rsph * invrdl2;                 // log2(I/i0), always <= 0

        // linearized exponent coefficients
        const float m = (2.0f * h2) * vlos;                 // vlab coefficient
        const float p = fmaf(-h2, vlos * vlos, cl2);        // constant term

        s_col[c * 160 + z] = make_float2(m, p);
    }
    __syncthreads();

    // ---- Phase 2: 10 waves, one 16-row frequency tile each ----
    const int wid  = tid >> 5;           // f-tile 0..9
    const int lane = tid & 31;
    const int frow = lane & 15;          // A-matrix row M = lane & 15
    const int khi  = (lane & 16) ? 8 : 0;// A-matrix K base per half-wave

    const int   ifine = wid * 16 + frow;
    const float ffine = ff0 + df * (float)ifine;
    const float vlab  = C_KMS * (F0_GHZ - ffine) / F0_GHZ - vshift;
    const float Kf    = h2 * vlab * vlab;   // per-row quadratic term (<= ~135)

    v8f acc = {0.f, 0.f, 0.f, 0.f, 0.f, 0.f, 0.f, 0.f};

    v16h bone;
#pragma unroll
    for (int j = 0; j < 16; ++j) bone[j] = (_Float16)1.0f;

    const float4* s4 = reinterpret_cast<const float4*>(s_col);

#pragma unroll
    for (int col = 0; col < 4; ++col) {          // 2x2 spatial pooling block
#pragma unroll
        for (int ch = 0; ch < 5; ++ch) {         // z in chunks of 32
            const int base = col * 160 + ch * 32;    // float2 index of chunk
            v16h a;
            // A 16-bit 16x32 layout: lane<16 -> K {0..7,16..23}; lane>=16 -> {8..15,24..31}
#pragma unroll
            for (int r = 0; r < 2; ++r) {
                const int kb  = base + khi + r * 16; // even -> float4 aligned
                const int f4b = kb >> 1;
#pragma unroll
                for (int jj = 0; jj < 4; ++jj) {
                    const float4 q = s4[f4b + jj];   // {m0,p0,m1,p1}
                    const float t0 = fmaf(vlab, q.x, q.y);
                    const float t1 = fmaf(vlab, q.z, q.w);
                    // argument (t - Kf) is always <= 0: raw v_exp_f32 is safe
                    // (underflow -> 0), no libm range-fixup needed.
                    const float e0 = __builtin_amdgcn_exp2f(t0 - Kf);
                    const float e1 = __builtin_amdgcn_exp2f(t1 - Kf);
                    a[r * 8 + jj * 2 + 0] = (_Float16)e0;
                    a[r * 8 + jj * 2 + 1] = (_Float16)e1;
                }
            }
            // D = A x ones + C : reduces the 16x32 weighted tile over z,
            // accumulating across z-chunks AND the 2x2 pooling block.
            acc = __builtin_amdgcn_wmma_f32_16x16x32_f16(
                false, a, false, bone, (short)0, acc, false, false);
        }
    }

    // ---- Phase 3: frequency pooling (x4) + scale, write coarse cube ----
    const float norm  = 1.0f / sqrtf(2.0f * 3.14159265358979323846f * s2);
    const float scale = i0 * norm * (1.0f / 16.0f);   // /16 = avg over 4x2x2 pool

    const int fbase = wid * 4;
    const int pix   = ic * 80 + jc;
    if (lane == 0) {            // rows 0..7 live in acc[0..7], column 0
        const float c0 = (acc[0] + acc[1] + acc[2] + acc[3]) * scale;
        const float c1 = (acc[4] + acc[5] + acc[6] + acc[7]) * scale;
        out[(fbase + 0) * 6400 + pix] = c0;
        out[(fbase + 1) * 6400 + pix] = c1;
    } else if (lane == 16) {    // rows 8..15
        const float c2 = (acc[0] + acc[1] + acc[2] + acc[3]) * scale;
        const float c3 = (acc[4] + acc[5] + acc[6] + acc[7]) * scale;
        out[(fbase + 2) * 6400 + pix] = c2;
        out[(fbase + 3) * 6400 + pix] = c3;
    }
}

extern "C" void kernel_launch(void* const* d_in, const int* in_sizes, int n_in,
                              void* d_out, int out_size, void* d_ws, size_t ws_size,
                              hipStream_t stream) {
    (void)in_sizes; (void)n_in; (void)out_size; (void)d_ws; (void)ws_size;
    ThickCube_wmma_kernel<<<dim3(6400), dim3(320), 0, stream>>>(
        (const float*)d_in[0],  // inclination
        (const float*)d_in[1],  // sky_rot
        (const float*)d_in[2],  // line_broadening
        (const float*)d_in[3],  // velocity_shift
        (const float*)d_in[4],  // v_max
        (const float*)d_in[5],  // r_turn
        (const float*)d_in[6],  // i0
        (const float*)d_in[7],  // r_d
        (const float*)d_in[8],  // freqs_coarse [40]
        (float*)d_out);         // [40,80,80] f32
}
